// SeqAttention_12695923327526
// MI455X (gfx1250) — matrix operand order repaired and checked by reading, not compiled, not run on hardware
//
#include <hip/hip_runtime.h>
#include <math.h>

#define BB 16
#define SS 2048
#define EE 512
#define HH 1024

typedef __attribute__((ext_vector_type(16))) _Float16 v16h;
typedef __attribute__((ext_vector_type(8)))  _Float16 v8h;
typedef __attribute__((ext_vector_type(4)))  _Float16 v4h;
typedef __attribute__((ext_vector_type(8)))  float    v8f;

// ---------------- WMMA fragment helpers (CDNA5 16x16x32 f16 layouts) ----------------
// A (16x32, MxK): lane 0-15 -> M=lane, elems 0..7 = K half*8..half*8+7, elems 8..15 = K 16+half*8..
__device__ __forceinline__ v16h load_a_frag(const _Float16* __restrict__ base, int ld,
                                            int kc, int lane) {
  const int row = lane & 15, half = lane >> 4;
  const _Float16* p = base + (size_t)row * ld + kc + half * 8;
  union { v16h v; v8h h[2]; } u;
  u.h[0] = *(const v8h*)(p);
  u.h[1] = *(const v8h*)(p + 16);
  return u.v;
}
// B (32x16, KxN): lane 0-15 -> N=lane, elems 0..7 = K half*8..half*8+7, elems 8..15 = K 16+half*8..
__device__ __forceinline__ v16h load_b_frag(const _Float16* __restrict__ base, int ld,
                                            int kc, int n0, int lane) {
  const int col = lane & 15, half = lane >> 4;
  const _Float16* p = base + (size_t)(kc + half * 8) * ld + n0 + col;
  v16h r;
#pragma unroll
  for (int s = 0; s < 8; ++s) {
    r[s] = p[(size_t)s * ld];
    r[s + 8] = p[(size_t)(s + 16) * ld];
  }
  return r;
}

// 32x64 per-wave tile: 2 M-subtiles x 4 N-subtiles, B-fragments reused across both A rows.
template <int KDIM>
__device__ __forceinline__ void gemm_32x64(const _Float16* __restrict__ A0, int lda,
                                           const _Float16* __restrict__ B, int ldb,
                                           int n0, int lane, v8f acc[2][4]) {
  const _Float16* A1 = A0 + (size_t)16 * lda;
  for (int kc = 0; kc < KDIM; kc += 32) {
    v16h a0 = load_a_frag(A0, lda, kc, lane);
    v16h a1 = load_a_frag(A1, lda, kc, lane);
#pragma unroll
    for (int t = 0; t < 4; ++t) {
      v16h bf = load_b_frag(B, ldb, kc, n0 + t * 16, lane);
      acc[0][t] = __builtin_amdgcn_wmma_f32_16x16x32_f16(false, a0, false, bf,
                                                         (short)0, acc[0][t], false, false);
      acc[1][t] = __builtin_amdgcn_wmma_f32_16x16x32_f16(false, a1, false, bf,
                                                         (short)0, acc[1][t], false, false);
    }
  }
}

// ---------------- conversions ----------------
__global__ __launch_bounds__(256) void cvt_kernel(const float* __restrict__ s,
                                                  _Float16* __restrict__ d,
                                                  float scale, int n) {
  int i = (blockIdx.x * 256 + threadIdx.x) * 4;
  if (i + 3 < n) {
    float4 f = *(const float4*)(s + i);
    v4h h;
    h[0] = (_Float16)(f.x * scale);
    h[1] = (_Float16)(f.y * scale);
    h[2] = (_Float16)(f.z * scale);
    h[3] = (_Float16)(f.w * scale);
    *(v4h*)(d + i) = h;
  }
}

// k[b][s][e] (f32) -> khT[b][e][s] (f16), 32x32 LDS tile transpose
__global__ __launch_bounds__(256) void ktrans_kernel(const float* __restrict__ k,
                                                     _Float16* __restrict__ khT) {
  __shared__ _Float16 tile[32][33];
  const int b = blockIdx.z;
  const int e0 = blockIdx.x << 5;
  const int s0 = blockIdx.y << 5;
  const int tx = threadIdx.x, ty = threadIdx.y;   // (32, 8)
  const float* src = k + (size_t)b * SS * EE;
  _Float16* dst = khT + (size_t)b * EE * SS;
#pragma unroll
  for (int j = 0; j < 4; ++j) {
    int s = s0 + ty + j * 8;
    tile[ty + j * 8][tx] = (_Float16)src[(size_t)s * EE + e0 + tx];
  }
  __syncthreads();
#pragma unroll
  for (int j = 0; j < 4; ++j) {
    int e = e0 + ty + j * 8;
    dst[(size_t)e * SS + s0 + tx] = tile[tx][ty + j * 8];
  }
}

// ---------------- scores = (q*scale) @ k^T  -> f16 ----------------
__global__ __launch_bounds__(128) void scores_gemm(const _Float16* __restrict__ qh,
                                                   const _Float16* __restrict__ khT,
                                                   _Float16* __restrict__ sp) {
  const int lane = threadIdx.x & 31;
  const int wave = threadIdx.x >> 5;
  const int b  = blockIdx.z;
  const int m0 = blockIdx.y << 5;                  // 32 query rows
  const int n0 = (blockIdx.x << 8) + (wave << 6);  // 64 key cols per wave
  const _Float16* Q  = qh  + (size_t)b * SS * EE + (size_t)m0 * EE;
  const _Float16* KT = khT + (size_t)b * EE * SS;
  _Float16* O = sp + (size_t)b * SS * SS;
  v8f acc[2][4] = {};
  gemm_32x64<EE>(Q, EE, KT, SS, n0, lane, acc);
  const int col = lane & 15, r0 = (lane >> 4) * 8;
#pragma unroll
  for (int i = 0; i < 2; ++i)
#pragma unroll
    for (int r = 0; r < 8; ++r) {
      size_t rowoff = (size_t)(m0 + i * 16 + r0 + r) * SS;
#pragma unroll
      for (int t = 0; t < 4; ++t)
        O[rowoff + n0 + t * 16 + col] = (_Float16)acc[i][t][r];
    }
}

// ---------------- row softmax over S, in-place f16 ----------------
__global__ __launch_bounds__(256) void softmax_kernel(_Float16* __restrict__ sp) {
  const size_t row = blockIdx.x;          // B*S rows
  _Float16* p = sp + row * SS;
  const int tid = threadIdx.x;
  __shared__ float red[256];
  float v[8];
  float m = -3.4e38f;
#pragma unroll
  for (int i = 0; i < 8; ++i) { v[i] = (float)p[tid + (i << 8)]; m = fmaxf(m, v[i]); }
  red[tid] = m; __syncthreads();
  for (int o = 128; o > 0; o >>= 1) { if (tid < o) red[tid] = fmaxf(red[tid], red[tid + o]); __syncthreads(); }
  m = red[0]; __syncthreads();
  float s = 0.f;
#pragma unroll
  for (int i = 0; i < 8; ++i) { v[i] = __expf(v[i] - m); s += v[i]; }
  red[tid] = s; __syncthreads();
  for (int o = 128; o > 0; o >>= 1) { if (tid < o) red[tid] += red[tid + o]; __syncthreads(); }
  float inv = 1.0f / red[0];
#pragma unroll
  for (int i = 0; i < 8; ++i) p[tid + (i << 8)] = (_Float16)(v[i] * inv);
}

// ---------------- attn_out = P @ V  -> f32 ----------------
__global__ __launch_bounds__(128) void pv_gemm(const _Float16* __restrict__ sp,
                                               const _Float16* __restrict__ vh,
                                               float* __restrict__ ao) {
  const int lane = threadIdx.x & 31;
  const int wave = threadIdx.x >> 5;
  const int b  = blockIdx.z;
  const int m0 = blockIdx.y << 5;
  const int n0 = (blockIdx.x << 8) + (wave << 6);
  const _Float16* P = sp + (size_t)b * SS * SS + (size_t)m0 * SS;
  const _Float16* V = vh + (size_t)b * SS * EE;
  float* O = ao + (size_t)b * SS * EE;
  v8f acc[2][4] = {};
  gemm_32x64<SS>(P, SS, V, EE, n0, lane, acc);
  const int col = lane & 15, r0 = (lane >> 4) * 8;
#pragma unroll
  for (int i = 0; i < 2; ++i)
#pragma unroll
    for (int r = 0; r < 8; ++r) {
      size_t rowoff = (size_t)(m0 + i * 16 + r0 + r) * EE;
#pragma unroll
      for (int t = 0; t < 4; ++t)
        O[rowoff + n0 + t * 16 + col] = acc[i][t][r];
    }
}

// ---------------- residual + LayerNorm ----------------
__global__ __launch_bounds__(256) void ln_kernel(const float* __restrict__ a,
                                                 const float* __restrict__ res,
                                                 const float* __restrict__ g,
                                                 const float* __restrict__ bt,
                                                 float* __restrict__ of,
                                                 _Float16* __restrict__ oh) {
  const int row = blockIdx.x;             // B*S rows, E=512, 2 elems/thread
  const int tid = threadIdx.x;
  __shared__ float red[256];
  const size_t base = (size_t)row * EE;
  float v0 = a[base + tid];
  float v1 = a[base + tid + 256];
  if (res) { v0 += res[base + tid]; v1 += res[base + tid + 256]; }
  red[tid] = v0 + v1; __syncthreads();
  for (int o = 128; o > 0; o >>= 1) { if (tid < o) red[tid] += red[tid + o]; __syncthreads(); }
  float mu = red[0] * (1.0f / EE); __syncthreads();
  float d0 = v0 - mu, d1 = v1 - mu;
  red[tid] = d0 * d0 + d1 * d1; __syncthreads();
  for (int o = 128; o > 0; o >>= 1) { if (tid < o) red[tid] += red[tid + o]; __syncthreads(); }
  float rs = rsqrtf(red[0] * (1.0f / EE) + 1e-5f);
  float o0 = d0 * rs * g[tid]       + bt[tid];
  float o1 = d1 * rs * g[tid + 256] + bt[tid + 256];
  of[base + tid] = o0;
  of[base + tid + 256] = o1;
  if (oh) {
    oh[base + tid] = (_Float16)o0;
    oh[base + tid + 256] = (_Float16)o1;
  }
}

// ---------------- h = gelu(x @ w1 + b1) -> f16 ----------------
__global__ __launch_bounds__(128) void ffn1_gemm(const _Float16* __restrict__ xh,
                                                 const _Float16* __restrict__ w1h,
                                                 const float* __restrict__ b1,
                                                 _Float16* __restrict__ hh) {
  const int lane = threadIdx.x & 31;
  const int wave = threadIdx.x >> 5;
  const int m0 = blockIdx.y << 5;                 // token tile over B*S
  const int n0 = (blockIdx.x << 8) + (wave << 6); // over H
  const _Float16* X = xh + (size_t)m0 * EE;
  v8f acc[2][4] = {};
  gemm_32x64<EE>(X, EE, w1h, HH, n0, lane, acc);
  const int col = lane & 15, r0 = (lane >> 4) * 8;
#pragma unroll
  for (int i = 0; i < 2; ++i)
#pragma unroll
    for (int r = 0; r < 8; ++r) {
      size_t rowoff = (size_t)(m0 + i * 16 + r0 + r) * HH;
#pragma unroll
      for (int t = 0; t < 4; ++t) {
        float x = acc[i][t][r] + b1[n0 + t * 16 + col];
        float gl = 0.5f * x * (1.0f + erff(x * 0.70710678118654752f));  // exact GELU
        hh[rowoff + n0 + t * 16 + col] = (_Float16)gl;
      }
    }
}

// ---------------- y = x + (h @ w2 + b2) -> f32 ----------------
__global__ __launch_bounds__(128) void ffn2_gemm(const _Float16* __restrict__ hh,
                                                 const _Float16* __restrict__ w2h,
                                                 const float* __restrict__ b2,
                                                 const float* __restrict__ xf,
                                                 float* __restrict__ y) {
  const int lane = threadIdx.x & 31;
  const int wave = threadIdx.x >> 5;
  const int m0 = blockIdx.y << 5;
  const int n0 = (blockIdx.x << 8) + (wave << 6); // over E
  const _Float16* Hm = hh + (size_t)m0 * HH;
  v8f acc[2][4] = {};
  gemm_32x64<HH>(Hm, HH, w2h, EE, n0, lane, acc);
  const int col = lane & 15, r0 = (lane >> 4) * 8;
#pragma unroll
  for (int i = 0; i < 2; ++i)
#pragma unroll
    for (int r = 0; r < 8; ++r) {
      size_t rowoff = (size_t)(m0 + i * 16 + r0 + r) * EE;
#pragma unroll
      for (int t = 0; t < 4; ++t) {
        size_t idx = rowoff + n0 + t * 16 + col;
        y[idx] = xf[idx] + acc[i][t][r] + b2[n0 + t * 16 + col];
      }
    }
}

extern "C" void kernel_launch(void* const* d_in, const int* in_sizes, int n_in,
                              void* d_out, int out_size, void* d_ws, size_t ws_size,
                              hipStream_t stream) {
  const float* q     = (const float*)d_in[0];
  const float* k     = (const float*)d_in[1];
  const float* v     = (const float*)d_in[2];
  const float* ln1_g = (const float*)d_in[3];
  const float* ln1_b = (const float*)d_in[4];
  const float* w1    = (const float*)d_in[5];
  const float* b1    = (const float*)d_in[6];
  const float* w2    = (const float*)d_in[7];
  const float* b2    = (const float*)d_in[8];
  const float* ln3_g = (const float*)d_in[9];
  const float* ln3_b = (const float*)d_in[10];
  float* out = (float*)d_out;

  char* ws = (char*)d_ws;
  size_t off = 0;
  auto carve = [&](size_t bytes) -> void* {
    void* p = ws + off;
    off = (off + bytes + 255) & ~(size_t)255;
    return p;
  };
  const size_t nqkv = (size_t)BB * SS * EE;   // 16.7M elems
  const size_t nw   = (size_t)EE * HH;        // 0.52M elems
  _Float16* qh  = (_Float16*)carve(nqkv * 2);
  _Float16* khT = (_Float16*)carve(nqkv * 2);
  _Float16* vh  = (_Float16*)carve(nqkv * 2);
  _Float16* w1h = (_Float16*)carve(nw * 2);
  _Float16* w2h = (_Float16*)carve(nw * 2);
  _Float16* sp  = (_Float16*)carve((size_t)BB * SS * SS * 2); // scores, then P in place
  float*    aoy = (float*)carve(nqkv * 4);    // attn_out, reused as y after LN1
  float*    xf  = (float*)carve(nqkv * 4);
  _Float16* xh  = (_Float16*)carve(nqkv * 2);
  _Float16* hh  = (_Float16*)carve((size_t)BB * SS * HH * 2);
  if (off > ws_size) return;  // workspace too small; nothing sensible to do

  const float scale = 0.04419417382415922f;   // 1/sqrt(512)

  cvt_kernel<<<(int)(nqkv / 1024), 256, 0, stream>>>(q, qh, scale, (int)nqkv);
  cvt_kernel<<<(int)(nqkv / 1024), 256, 0, stream>>>(v, vh, 1.0f, (int)nqkv);
  cvt_kernel<<<(int)(nw   / 1024), 256, 0, stream>>>(w1, w1h, 1.0f, (int)nw);
  cvt_kernel<<<(int)(nw   / 1024), 256, 0, stream>>>(w2, w2h, 1.0f, (int)nw);
  ktrans_kernel<<<dim3(EE / 32, SS / 32, BB), dim3(32, 8), 0, stream>>>(k, khT);

  scores_gemm<<<dim3(SS / 256, SS / 32, BB), 128, 0, stream>>>(qh, khT, sp);
  softmax_kernel<<<BB * SS, 256, 0, stream>>>(sp);
  pv_gemm<<<dim3(EE / 256, SS / 32, BB), 128, 0, stream>>>(sp, vh, aoy);

  ln_kernel<<<BB * SS, 256, 0, stream>>>(aoy, q, ln1_g, ln1_b, xf, xh);

  ffn1_gemm<<<dim3(HH / 256, (BB * SS) / 32), 128, 0, stream>>>(xh, w1h, b1, hh);
  ffn2_gemm<<<dim3(EE / 256, (BB * SS) / 32), 128, 0, stream>>>(hh, w2h, b2, xf, aoy);

  ln_kernel<<<BB * SS, 256, 0, stream>>>(aoy, nullptr, ln3_g, ln3_b, out, nullptr);
}
